// CrossAttention_23081154248955
// MI455X (gfx1250) — matrix operand order repaired and checked by reading, not compiled, not run on hardware
//
#include <hip/hip_runtime.h>

// CDNA5 (gfx1250) bf16 WMMA cross-attention, flash-style.
// ws layout (bf16): Q [4,16,2048,64] @0, K [4,16,1024,64] @16MB, Vt [4,16,64,1024] @24MB.

typedef __attribute__((ext_vector_type(16))) __bf16 v16bf;
typedef __attribute__((ext_vector_type(8)))  __bf16 v8bf;
typedef __attribute__((ext_vector_type(8)))  float  v8f;

static __device__ __forceinline__ v8f wmma_bf16(v16bf a, v16bf b, v8f c) {
  // (neg_a, A, neg_b, B, c_mod, C, reuse_a, reuse_b)
  return __builtin_amdgcn_wmma_f32_16x16x32_bf16(false, a, false, b, (short)0, c, false, false);
}

static __device__ __forceinline__ v16bf cat8(v8bf lo, v8bf hi) {
  union { v16bf v; v8bf h[2]; } u;
  u.h[0] = lo; u.h[1] = hi;
  return u.v;
}

// Build a 16x32 bf16 A-fragment from an fp32 row (on-the-fly downconvert).
// lanes 0-15: K = 0..15; lanes 16-31: K = 16..31 (contiguous, matches B).
static __device__ __forceinline__ v16bf a_frag_f32(const float* row, int k0, int hi) {
  v16bf a;
  const float* a1 = row + k0 + hi * 16;
  const float* a2 = row + k0 + 8 + hi * 16;
#pragma unroll
  for (int i = 0; i < 8; ++i) { a[i] = (__bf16)a1[i]; a[8 + i] = (__bf16)a2[i]; }
  return a;
}

// Build a 32x16 bf16 B-fragment from an fp32 row: 16 contiguous K per half-wave.
static __device__ __forceinline__ v16bf b_frag_f32(const float* row, int k0, int hi) {
  v16bf b;
  const float* bp = row + k0 + hi * 16;
#pragma unroll
  for (int i = 0; i < 16; ++i) b[i] = (__bf16)bp[i];
  return b;
}

// ---------------- Q projection: Q[m,c] = sum_k x[m,k]*Wq[c,k]
// 32x64 tile per wave: 2 A-frags x 4 B-frags -> 8 WMMA per 32-wide k-step.
__global__ __launch_bounds__(128) void ca_qproj(const float* __restrict__ x,
                                                const float* __restrict__ Wq,
                                                __bf16* __restrict__ q_ws) {
  const int lane = threadIdx.x & 31;
  const int wave = threadIdx.x >> 5;
  const int tile = blockIdx.x * 4 + wave;   // 0..4095 = 256 row-tiles x 16 col-tiles
  const int rt = tile >> 4;
  const int ct = tile & 15;
  const int l15 = lane & 15;
  const int hi  = lane >> 4;

  const float* xrow0 = x + (size_t)(rt * 32 + l15) * 1024;
  const float* xrow1 = x + (size_t)(rt * 32 + 16 + l15) * 1024;
  const float* wrow[4];
#pragma unroll
  for (int n = 0; n < 4; ++n) wrow[n] = Wq + (size_t)(ct * 64 + n * 16 + l15) * 1024;

  v8f acc[2][4] = {};
  for (int ks = 0; ks < 32; ++ks) {
    const int k0 = ks * 32;
    v16bf a0 = a_frag_f32(xrow0, k0, hi);
    v16bf a1 = a_frag_f32(xrow1, k0, hi);
    v16bf b[4];
#pragma unroll
    for (int n = 0; n < 4; ++n) b[n] = b_frag_f32(wrow[n], k0, hi);
#pragma unroll
    for (int n = 0; n < 4; ++n) {
      acc[0][n] = wmma_bf16(a0, b[n], acc[0][n]);
      acc[1][n] = wmma_bf16(a1, b[n], acc[1][n]);
    }
  }

#pragma unroll
  for (int r = 0; r < 2; ++r) {
#pragma unroll
    for (int n = 0; n < 4; ++n) {
      const int col = ct * 64 + n * 16 + l15;
      const int h = col >> 6, dd = col & 63;
#pragma unroll
      for (int j = 0; j < 8; ++j) {
        const int row = rt * 32 + r * 16 + j + 8 * hi;   // D: row = vgpr + 8*(lane>=16)
        const int bb = row >> 11, nn = row & 2047;
        q_ws[(((size_t)bb * 16 + h) * 2048 + nn) * 64 + dd] = (__bf16)acc[r][n][j];
      }
    }
  }
}

// ---------------- KV projection: KV[m,j] = sum_k y[m,k]*Wkv[j,k]; 32x64 tile per wave.
__global__ __launch_bounds__(128) void ca_kvproj(const float* __restrict__ y,
                                                 const float* __restrict__ Wkv,
                                                 __bf16* __restrict__ k_ws,
                                                 __bf16* __restrict__ vt_ws) {
  const int lane = threadIdx.x & 31;
  const int wave = threadIdx.x >> 5;
  const int tile = blockIdx.x * 4 + wave;   // 0..4095 = 128 row-tiles x 32 col-tiles
  const int rt = tile >> 5;
  const int ct = tile & 31;
  const int l15 = lane & 15;
  const int hi  = lane >> 4;

  const float* yrow0 = y + (size_t)(rt * 32 + l15) * 768;
  const float* yrow1 = y + (size_t)(rt * 32 + 16 + l15) * 768;
  const float* wrow[4];
#pragma unroll
  for (int n = 0; n < 4; ++n) wrow[n] = Wkv + (size_t)(ct * 64 + n * 16 + l15) * 768;

  v8f acc[2][4] = {};
  for (int ks = 0; ks < 24; ++ks) {
    const int k0 = ks * 32;
    v16bf a0 = a_frag_f32(yrow0, k0, hi);
    v16bf a1 = a_frag_f32(yrow1, k0, hi);
    v16bf b[4];
#pragma unroll
    for (int n = 0; n < 4; ++n) b[n] = b_frag_f32(wrow[n], k0, hi);
#pragma unroll
    for (int n = 0; n < 4; ++n) {
      acc[0][n] = wmma_bf16(a0, b[n], acc[0][n]);
      acc[1][n] = wmma_bf16(a1, b[n], acc[1][n]);
    }
  }

#pragma unroll
  for (int r = 0; r < 2; ++r) {
#pragma unroll
    for (int n = 0; n < 4; ++n) {
      const int col = ct * 64 + n * 16 + l15;   // j in [0,2048): <1024 -> K, >=1024 -> Vt
#pragma unroll
      for (int j = 0; j < 8; ++j) {
        const int row = rt * 32 + r * 16 + j + 8 * hi;
        const int bb = row >> 10, n2 = row & 1023;
        const __bf16 val = (__bf16)acc[r][n][j];
        if (col < 1024) {
          const int h = col >> 6, dd = col & 63;
          k_ws[(((size_t)bb * 16 + h) * 1024 + n2) * 64 + dd] = val;
        } else {
          const int jv = col - 1024;
          const int h = jv >> 6, dd = jv & 63;
          vt_ws[(((size_t)bb * 16 + h) * 64 + dd) * 1024 + n2] = val;   // Vt[b,h,d,N2]
        }
      }
    }
  }
}

// ---------------- Flash attention: one wave per 16-query tile
struct WaveLds {
  float  sc[16][32];     // raw scores for current 32-key chunk
  __bf16 p[16][32];      // exp'ed probabilities (bf16 A operand), 64B rows
  float  alpha[16];      // per-row rescale factor
  float  lsum[16];       // per-row final denominators
};

__global__ __launch_bounds__(128) void ca_attn(const __bf16* __restrict__ q_ws,
                                               const __bf16* __restrict__ k_ws,
                                               const __bf16* __restrict__ vt_ws,
                                               const unsigned char* __restrict__ pad_mask,
                                               float* __restrict__ out) {
  __shared__ WaveLds lds[4];
  const int lane = threadIdx.x & 31;
  const int wave = threadIdx.x >> 5;
  WaveLds& L = lds[wave];

  const int tile = blockIdx.x * 4 + wave;   // 0..8191 = 64 (b,h) x 128 q-tiles
  const int bh = tile >> 7;
  const int qt = tile & 127;
  const int b = bh >> 4, h = bh & 15;
  const int qbase = qt * 16;
  const int l15 = lane & 15;
  const int hi  = lane >> 4;

  // Q A-fragments (d = 0..31 and 32..63) as vector loads; scale folded into softmax.
  const v8bf* q8 = (const v8bf*)(q_ws + ((size_t)bh * 2048 + (qbase + l15)) * 64);
  const v16bf aq0 = cat8(q8[2 * hi], q8[2 * hi + 1]);
  const v16bf aq1 = cat8(q8[4 + 2 * hi], q8[5 + 2 * hi]);

  // Query-axis pad mask: masked row -> constant scores -> uniform softmax (matches reference).
  float keep = 1.0f;
  if (lane < 16) keep = pad_mask[(size_t)b * 2048 + qbase + lane] ? 1.0f : 0.0f;

  v8f o[4] = {};                 // 16x64 output accumulator (4 col tiles)
  float m_run = -1e30f, l_run = 0.0f;

  const __bf16* kbase_p = k_ws + (size_t)bh * 1024 * 64;

  for (int kc = 0; kc < 32; ++kc) {
    const int kb = kc * 32;

    // Prefetch next chunk's K rows while this chunk computes (global_prefetch_b8).
    if (kc + 1 < 32) {
      __builtin_prefetch(kbase_p + (size_t)(kb + 32 + lane) * 64, 0, 0);
      __builtin_prefetch(vt_ws + ((size_t)bh * 64 + lane) * 1024 + kb + 32, 0, 0);
    }

    // --- scores: S = (Q * scale) . K^T for 32 keys (two 16-key subtiles)
    const v8bf* k08 = (const v8bf*)(kbase_p + (size_t)(kb + l15) * 64);
    const v8bf* k18 = (const v8bf*)(kbase_p + (size_t)(kb + 16 + l15) * 64);
    const v16bf bk00 = cat8(k08[2 * hi], k08[2 * hi + 1]);       // d 0..31
    const v16bf bk01 = cat8(k08[4 + 2 * hi], k08[5 + 2 * hi]);   // d 32..63
    const v16bf bk10 = cat8(k18[2 * hi], k18[2 * hi + 1]);
    const v16bf bk11 = cat8(k18[4 + 2 * hi], k18[5 + 2 * hi]);

    v8f s_lo = {}, s_hi = {};
    s_lo = wmma_bf16(aq0, bk00, s_lo);
    s_lo = wmma_bf16(aq1, bk01, s_lo);
    s_hi = wmma_bf16(aq0, bk10, s_hi);
    s_hi = wmma_bf16(aq1, bk11, s_hi);

    // D-layout -> LDS (row = vgpr + 8*hi, col = l15)
#pragma unroll
    for (int j = 0; j < 8; ++j) {
      L.sc[j + 8 * hi][l15]      = s_lo[j];
      L.sc[j + 8 * hi][16 + l15] = s_hi[j];
    }
    asm volatile("s_wait_dscnt 0" ::: "memory");

    // --- online softmax, one row per lane (lanes 0..15)
    if (lane < 16) {
      const int r = lane;
      float s[32];
      float mx = m_run;
#pragma unroll
      for (int i = 0; i < 32; ++i) {
        const float v = 0.125f * L.sc[r][i] * keep;   // scale = d^-0.5; masked row -> 0
        s[i] = v;
        mx = fmaxf(mx, v);
      }
      const float alpha = __expf(m_run - mx);
      float sum = 0.0f;
#pragma unroll
      for (int i = 0; i < 32; ++i) {
        const float p = __expf(s[i] - mx);
        sum += p;
        L.p[r][i] = (__bf16)p;
      }
      L.alpha[r] = alpha;
      m_run = mx;
      l_run = l_run * alpha + sum;
    }
    asm volatile("s_wait_dscnt 0" ::: "memory");

    // --- rescale accumulators by per-row alpha
#pragma unroll
    for (int j = 0; j < 8; ++j) {
      const float a = L.alpha[j + 8 * hi];
      o[0][j] *= a; o[1][j] *= a; o[2][j] *= a; o[3][j] *= a;
    }

    // --- P as A-fragment (K = 32 keys of this chunk), vector LDS reads
    const v8bf* p8 = (const v8bf*)&L.p[l15][0];
  const v16bf ap = cat8(p8[2 * hi], p8[2 * hi + 1]);

    // --- O += P . V ; Vt[b,h,d,N2] gives contiguous key runs per lane
#pragma unroll
    for (int n = 0; n < 4; ++n) {
      const v8bf* v8p = (const v8bf*)(vt_ws + ((size_t)bh * 64 + (n * 16 + l15)) * 1024 + kb);
      const v16bf bv = cat8(v8p[2 * hi], v8p[2 * hi + 1]);
      o[n] = wmma_bf16(ap, bv, o[n]);
    }
  }

  if (lane < 16) L.lsum[lane] = l_run;
  asm volatile("s_wait_dscnt 0" ::: "memory");

#pragma unroll
  for (int j = 0; j < 8; ++j) {
    const int r = j + 8 * hi;
    const float inv = 1.0f / L.lsum[r];
    const int qrow = qbase + r;
    float* orow = out + ((size_t)b * 2048 + qrow) * 1024 + h * 64;
#pragma unroll
    for (int n = 0; n < 4; ++n) orow[n * 16 + l15] = o[n][j] * inv;
  }
}

extern "C" void kernel_launch(void* const* d_in, const int* in_sizes, int n_in,
                              void* d_out, int out_size, void* d_ws, size_t ws_size,
                              hipStream_t stream) {
  const float*         x   = (const float*)d_in[0];          // [4,2048,1024] f32
  const float*         y   = (const float*)d_in[1];          // [4,1024,768]  f32
  const unsigned char* pad = (const unsigned char*)d_in[2];  // [4,2048] bool
  const float*         Wq  = (const float*)d_in[3];          // [1024,1024] f32
  const float*         Wkv = (const float*)d_in[4];          // [2048,768]  f32
  float* out = (float*)d_out;                                // [4,2048,1024] f32

  __bf16* q_ws  = (__bf16*)d_ws;                        // 8M elems (16 MB)
  __bf16* k_ws  = q_ws + (size_t)8 * 1024 * 1024;       // 4M elems (8 MB)
  __bf16* vt_ws = k_ws + (size_t)4 * 1024 * 1024;       // 4M elems (8 MB)

  ca_qproj<<<1024, 128, 0, stream>>>(x, Wq, q_ws);
  ca_kvproj<<<1024, 128, 0, stream>>>(y, Wkv, k_ws, vt_ws);
  ca_attn<<<2048, 128, 0, stream>>>(q_ws, k_ws, vt_ws, pad, out);
}
